// MamboPoolingWithEdgeGraphScore_37572373905742
// MI455X (gfx1250) — compile-verified, matrix operand order checked by reading, not run on hardware
//
#include <hip/hip_runtime.h>
#include <math.h>

// ---------------- problem constants (from reference) ----------------
#define N_NODES   20000
#define E_DIR     160000     // directed edges in edge_index (2 x 160000)
#define E_MAX     80000      // max unique undirected edges
#define IN_CH     128
#define HID       64
#define EPS_BN    1e-5f
#define HASH1_CAP 262144
#define HASH2_CAP 524288
#define HASH_EMPTY 0xFFFFFFFFFFFFFFFFull

// ---------------- WMMA / TDM vector types (gfx1250, wave32) ----------------
typedef __bf16 bf16_t;
typedef __attribute__((ext_vector_type(16))) __bf16 v16bf;
typedef __attribute__((ext_vector_type(8)))  float  v8f;
typedef __attribute__((ext_vector_type(4)))  unsigned int u32x4;
typedef __attribute__((ext_vector_type(8)))  int          i32x8;
typedef __attribute__((ext_vector_type(4)))  int          i32x4;

union FragBF16 { uint4 q[2]; v16bf v; };

__device__ __forceinline__ unsigned short f2bf(float f) {
  unsigned int u = __float_as_uint(f);
  unsigned int lsb = (u >> 16) & 1u;
  u += 0x7fffu + lsb;                       // round-to-nearest-even
  return (unsigned short)(u >> 16);
}

// Issue one TDM 2-D tile load into LDS (wave-level; EXEC-independent).
// rows beyond tensor_dim1 read as zero per ISA OOB rule.
__device__ __forceinline__ void tdm_load_2d(unsigned lds_addr, unsigned long long gaddr,
                                            unsigned dim0_dw, unsigned rows_tensor,
                                            unsigned tile_rows, unsigned stride_dw,
                                            unsigned pad_interval_code, unsigned pad_amount_code) {
  // group0: count=1 | lds_addr | global_addr[56:0] | type=2
  u32x4 g0 = { 1u, lds_addr, (unsigned)gaddr,
               (unsigned)((gaddr >> 32) & 0x01FFFFFFu) | 0x80000000u };
  // group1: data_size=4B(code2), pad_enable, pad interval/amount, dims/tiles/stride
  unsigned d0 = (2u << 16) | (1u << 20) | (pad_interval_code << 22) | (pad_amount_code << 25);
  unsigned d1 = (dim0_dw & 0xFFFFu) << 16;                       // tensor_dim0[15:0] @63:48
  unsigned d2 = (dim0_dw >> 16) | ((rows_tensor & 0xFFFFu) << 16); // dim0 hi | dim1 lo
  unsigned d3 = (rows_tensor >> 16) | (dim0_dw << 16);           // dim1 hi | tile_dim0
  unsigned d4 = tile_rows;                                       // tile_dim1 | tile_dim2=0
  unsigned d5 = stride_dw;                                       // tensor_dim0_stride lo
  i32x8 g1 = { (int)d0, (int)d1, (int)d2, (int)d3, (int)d4, (int)d5, 0, 0 };
  i32x4 z4 = { 0, 0, 0, 0 };
#if __clang_major__ >= 23
  i32x8 z8 = { 0, 0, 0, 0, 0, 0, 0, 0 };
  __builtin_amdgcn_tensor_load_to_lds(g0, g1, z4, z4, z8, 0);
#else
  __builtin_amdgcn_tensor_load_to_lds(g0, g1, z4, z4, 0);
#endif
}

// ---------------- 1) undirected-edge dedup + node degrees ----------------
__global__ void k_dedup(const int* __restrict__ ei, unsigned long long* __restrict__ tbl,
                        int* __restrict__ csrc, int* __restrict__ cdst,
                        int* __restrict__ deg, int* __restrict__ ctr) {
  int i = blockIdx.x * 256 + threadIdx.x;
  if (i >= E_DIR) return;
  int s = ei[i], d = ei[E_DIR + i];
  if (s >= d) return;
  unsigned long long key = (unsigned long long)s * (unsigned long long)N_NODES + (unsigned long long)d;
  unsigned int slot = (unsigned int)((key * 0x9E3779B97F4A7C15ull) >> 46) & (HASH1_CAP - 1);
  for (;;) {
    unsigned long long prev = atomicCAS(&tbl[slot], HASH_EMPTY, key);
    if (prev == HASH_EMPTY) {
      int idx = atomicAdd(&ctr[0], 1);
      csrc[idx] = s; cdst[idx] = d;
      atomicAdd(&deg[s], 1); atomicAdd(&deg[d], 1);
      return;
    }
    if (prev == key) return;
    slot = (slot + 1) & (HASH1_CAP - 1);
  }
}

__global__ void k_dinv(const int* __restrict__ csrc, const int* __restrict__ cdst,
                       const int* __restrict__ deg, float* __restrict__ dinv,
                       const int* __restrict__ ctr) {
  int e = blockIdx.x * 256 + threadIdx.x;
  if (e >= ctr[0]) return;
  int dl = deg[csrc[e]] + deg[cdst[e]] - 1;
  if (dl < 1) dl = 1;
  dinv[e] = rsqrtf((float)dl);
}

// ---------------- 2) line_x = 0.5*(x[u]+x[v]) packed bf16 (row-major) ----------------
__global__ void k_linex(const float* __restrict__ x, const int* __restrict__ csrc,
                        const int* __restrict__ cdst, unsigned int* __restrict__ A,
                        const int* __restrict__ ctr) {
  int idx = blockIdx.x * 256 + threadIdx.x;
  int e = idx >> 6, p = idx & 63;
  if (e >= E_MAX) return;
  unsigned int w = 0u;
  if (e < ctr[0]) {
    int u = csrc[e], v = cdst[e];
    const float* xu = x + (size_t)u * IN_CH;
    const float* xv = x + (size_t)v * IN_CH;
    __builtin_prefetch(xu, 0, 3);
    float a0 = 0.5f * (xu[2 * p]     + xv[2 * p]);
    float a1 = 0.5f * (xu[2 * p + 1] + xv[2 * p + 1]);
    w = (unsigned int)f2bf(a0) | ((unsigned int)f2bf(a1) << 16);
  }
  A[(size_t)e * 64 + p] = w;
}

// W1 -> bf16, pre-swizzled into per-lane fragment-major order.
// fragment f = nt*128 + kc*32 + half*16 + ncol ; 16 bf16 per fragment:
// elements 0..7  -> K = kc*32 + 8*half + 0..7
// elements 8..15 -> K = kc*32 + 8*half + 16..23 ; column = nt*16 + ncol
__global__ void k_w1bf(const float* __restrict__ W1, unsigned int* __restrict__ B) {
  int j = blockIdx.x * 256 + threadIdx.x;          // 4096 packed uints (16 KB)
  if (j >= 4096) return;
  int frag = j >> 3, w = j & 7;
  int ncol = frag & 15, half = (frag >> 4) & 1, kc = (frag >> 5) & 3, nt = (frag >> 7) & 3;
  int e0 = 2 * w, e1 = 2 * w + 1;
  int k0 = kc * 32 + 8 * half + (e0 < 8 ? e0 : 8 + e0);
  int k1 = kc * 32 + 8 * half + (e1 < 8 ? e1 : 8 + e1);
  int col = nt * 16 + ncol;
  B[j] = (unsigned int)f2bf(W1[k0 * HID + col]) |
         ((unsigned int)f2bf(W1[k1 * HID + col]) << 16);
}

// ---------------- 3) GEMM: H = line_x(E x128) @ W1(128x64), bf16 WMMA ----------------
// TDM stages both operands into LDS (with pad-insertion for bank-conflict-free
// b128 fragment loads); 8 waves x (1 M-tile x 4 N-tiles) per block.
__global__ __launch_bounds__(256) void k_gemm1(const unsigned int* __restrict__ Abf,
                                               const unsigned int* __restrict__ Bbf,
                                               float* __restrict__ H,
                                               const int* __restrict__ ctr) {
  // A: 128 rows x 64 dwords, +4 dwords pad per row -> 272B row stride (34 KB)
  __shared__ __align__(16) unsigned char Alds[128 * 272];
  // B: 512 fragments x 8 dwords, +4 dwords pad per fragment -> 48B stride (24 KB)
  __shared__ __align__(16) unsigned char Blds[512 * 48];
  const int Ec    = ctr[0];
  const int tid   = threadIdx.x;
  const int ebase = blockIdx.x * 128;

  if (tid < 32) {                               // wave 0 drives the TDM
    unsigned rows = (Ec > ebase) ? (unsigned)(Ec - ebase) : 0u;
    tdm_load_2d((unsigned)(uintptr_t)&Alds[0],
                (unsigned long long)(uintptr_t)(Abf + (size_t)ebase * 64),
                /*dim0_dw=*/64u, /*rows_tensor=*/rows, /*tile_rows=*/128u,
                /*stride_dw=*/64u, /*pad_interval(64dw)=*/5u, /*pad_amount(4dw)=*/3u);
    tdm_load_2d((unsigned)(uintptr_t)&Blds[0],
                (unsigned long long)(uintptr_t)Bbf,
                /*dim0_dw=*/8u, /*rows_tensor=*/512u, /*tile_rows=*/512u,
                /*stride_dw=*/8u, /*pad_interval(8dw)=*/2u, /*pad_amount(4dw)=*/3u);
    __builtin_amdgcn_s_wait_tensorcnt(0);
  }
  __syncthreads();

  const int wave  = tid >> 5;
  const int lane  = tid & 31;
  const int half  = lane >> 4;
  const int mlane = lane & 15;

  const unsigned char* arow = Alds + (size_t)(wave * 16 + mlane) * 272;

  v8f acc[4] = {};
#pragma unroll
  for (int kc = 0; kc < 4; ++kc) {
    FragBF16 a;                                  // two contiguous b128 runs
    a.q[0] = *(const uint4*)(arow + kc * 64 + half * 16);
    a.q[1] = *(const uint4*)(arow + kc * 64 + 32 + half * 16);
#pragma unroll
    for (int nt = 0; nt < 4; ++nt) {
      int f = nt * 128 + kc * 32 + half * 16 + mlane;
      FragBF16 b;
      b.q[0] = *(const uint4*)(Blds + (size_t)f * 48);
      b.q[1] = *(const uint4*)(Blds + (size_t)f * 48 + 16);
      acc[nt] = __builtin_amdgcn_wmma_f32_16x16x32_bf16(
          false, a.v, false, b.v, (short)0, acc[nt], false, false);
    }
  }
  const int rowb = ebase + wave * 16;
#pragma unroll
  for (int nt = 0; nt < 4; ++nt)
#pragma unroll
    for (int v = 0; v < 8; ++v) {               // C/D: M = v + 8*half, N = lane&15
      int m = rowb + v + 8 * half;
      if (m < Ec) H[(size_t)m * HID + nt * 16 + mlane] = acc[nt][v];
    }
}

// ---------------- 4) node accumulators S(n) = sum_{f at n} h[f]*dinv[f] ----------------
__global__ void k_scatterS(const float* __restrict__ H, const float* __restrict__ dinv,
                           const int* __restrict__ csrc, const int* __restrict__ cdst,
                           float* __restrict__ S, const int* __restrict__ ctr) {
  int idx = blockIdx.x * 256 + threadIdx.x;
  int e = idx >> 6, c = idx & 63;
  if (e >= ctr[0]) return;
  float hv = H[idx] * dinv[e];
  atomicAdd(&S[(size_t)csrc[e] * HID + c], hv);
  atomicAdd(&S[(size_t)cdst[e] * HID + c], hv);
}

__global__ void k_conv1(float* __restrict__ H, const float* __restrict__ S,
                        const float* __restrict__ dinv, const int* __restrict__ csrc,
                        const int* __restrict__ cdst, const float* __restrict__ b1,
                        float* __restrict__ bnacc, const int* __restrict__ ctr) {
  __shared__ float ssum[256], ssq[256];
  int tid = threadIdx.x;
  int idx = blockIdx.x * 256 + tid;
  int e = idx >> 6, c = idx & 63;
  float val = 0.f;
  bool ok = (e < ctr[0]);
  if (ok) {
    float di = dinv[e];
    int u = csrc[e], v = cdst[e];
    val = di * (S[(size_t)u * HID + c] + S[(size_t)v * HID + c]) - di * di * H[idx] + b1[c];
    H[idx] = val;
  }
  ssum[tid] = ok ? val : 0.f;
  ssq[tid]  = ok ? val * val : 0.f;
  __syncthreads();
  if (tid < 64) {
    float s = ssum[tid] + ssum[tid + 64] + ssum[tid + 128] + ssum[tid + 192];
    float q = ssq[tid]  + ssq[tid + 64]  + ssq[tid + 128]  + ssq[tid + 192];
    atomicAdd(&bnacc[tid], s);
    atomicAdd(&bnacc[64 + tid], q);
  }
}

__global__ void k_bn_relu(float* __restrict__ H, const float* __restrict__ bnacc,
                          const float* __restrict__ gamma, const float* __restrict__ beta,
                          const int* __restrict__ ctr) {
  int idx = blockIdx.x * 256 + threadIdx.x;
  int e = idx >> 6, c = idx & 63;
  int Ec = ctr[0];
  if (e >= Ec) return;
  float inv = 1.0f / (float)(Ec > 0 ? Ec : 1);
  float mu = bnacc[c] * inv;
  float var = bnacc[64 + c] * inv - mu * mu;
  float h = gamma[c] * (H[idx] - mu) * rsqrtf(var + EPS_BN) + beta[c];
  H[idx] = fmaxf(h, 0.f);
}

// ---------------- 5) second conv (64 -> 1) + sigmoid score + histogram ----------------
__global__ void k_gemv2(const float* __restrict__ H, const float* __restrict__ W2,
                        float* __restrict__ val, const int* __restrict__ ctr) {
  int e = blockIdx.x * 256 + threadIdx.x;
  if (e >= ctr[0]) return;
  const float* h = H + (size_t)e * HID;
  float acc = 0.f;
#pragma unroll
  for (int c = 0; c < HID; ++c) acc += h[c] * W2[c];
  val[e] = acc;
}

__global__ void k_scatterS2(const float* __restrict__ val, const float* __restrict__ dinv,
                            const int* __restrict__ csrc, const int* __restrict__ cdst,
                            float* __restrict__ S2, const int* __restrict__ ctr) {
  int e = blockIdx.x * 256 + threadIdx.x;
  if (e >= ctr[0]) return;
  float hv = val[e] * dinv[e];
  atomicAdd(&S2[csrc[e]], hv);
  atomicAdd(&S2[cdst[e]], hv);
}

__global__ void k_score(const float* __restrict__ val, const float* __restrict__ dinv,
                        const int* __restrict__ csrc, const int* __restrict__ cdst,
                        const float* __restrict__ S2, const float* __restrict__ b2,
                        float* __restrict__ scores, int* __restrict__ hist,
                        const int* __restrict__ ctr) {
  int e = blockIdx.x * 256 + threadIdx.x;
  if (e >= ctr[0]) return;
  float di = dinv[e];
  float z = di * (S2[csrc[e]] + S2[cdst[e]]) - di * di * val[e] + b2[0];
  float s = 1.f / (1.f + __expf(-z));
  scores[e] = s;
  int b = (int)(s * 1024.f);
  if (b < 0) b = 0; if (b > 1023) b = 1023;
  atomicAdd(&hist[b], 1);
}

__global__ void k_threshold(const int* __restrict__ hist, int* __restrict__ ctr) {
  int Ec = ctr[0];
  int k = Ec / 2;
  if (k > N_NODES / 2) k = N_NODES / 2;
  if (k < 1) k = 1;
  int cum = 0, T = -1, rem = 0;
  for (int b = 1023; b >= 0; --b) {
    int nb = hist[b];
    if (cum + nb >= k) { T = b; rem = k - cum; break; }
    cum += nb;
  }
  if (T < 0) { T = 0; rem = k - cum; }
  ctr[4] = T; ctr[5] = rem;
}

// ---------------- 6) clustering ----------------
__global__ void k_n2c_init(int* __restrict__ n2c) {
  int n = blockIdx.x * 256 + threadIdx.x;
  if (n < N_NODES) n2c[n] = n;
}

__global__ void k_select(const float* __restrict__ scores, const int* __restrict__ csrc,
                         const int* __restrict__ cdst, int* __restrict__ n2c,
                         int* __restrict__ ctr) {
  int e = blockIdx.x * 256 + threadIdx.x;
  if (e >= ctr[0]) return;
  float s = scores[e];
  int b = (int)(s * 1024.f);
  if (b < 0) b = 0; if (b > 1023) b = 1023;
  int T = ctr[4], rem = ctr[5];
  bool sel = (b > T) || (b == T && atomicAdd(&ctr[1], 1) < rem);
  if (sel) atomicExch(&n2c[cdst[e]], csrc[e]);
}

__global__ void k_present(const int* __restrict__ n2c, int* __restrict__ present) {
  int n = blockIdx.x * 256 + threadIdx.x;
  if (n < N_NODES) present[n2c[n]] = 1;
}

__global__ void k_scan1(const int* __restrict__ present, int* __restrict__ part,
                        int* __restrict__ bsums) {
  __shared__ int sh[256];
  int tid = threadIdx.x;
  int g = blockIdx.x * 256 + tid;
  int v = (g < N_NODES) ? present[g] : 0;
  sh[tid] = v;
  __syncthreads();
  for (int off = 1; off < 256; off <<= 1) {
    int t = (tid >= off) ? sh[tid - off] : 0;
    __syncthreads();
    sh[tid] += t;
    __syncthreads();
  }
  if (g < N_NODES) part[g] = sh[tid] - v;
  if (tid == 255) bsums[blockIdx.x] = sh[255];
}

__global__ void k_scan2(int* __restrict__ bsums, int nblocks, int* __restrict__ ctr) {
  if (threadIdx.x != 0 || blockIdx.x != 0) return;
  int acc = 0;
  for (int i = 0; i < nblocks; ++i) { int t = bsums[i]; bsums[i] = acc; acc += t; }
  ctr[3] = acc;
}

__global__ void k_scan3(const int* __restrict__ present, int* __restrict__ part,
                        const int* __restrict__ bsums, int* __restrict__ uniq) {
  int g = blockIdx.x * 256 + threadIdx.x;
  if (g >= N_NODES) return;
  int rank = part[g] + bsums[g >> 8];
  part[g] = rank;
  if (present[g]) uniq[rank] = g;
}

__global__ void k_nodemap(const int* __restrict__ n2c, const int* __restrict__ o2n,
                          int* __restrict__ nodemap) {
  int n = blockIdx.x * 256 + threadIdx.x;
  if (n < N_NODES) nodemap[n] = o2n[n2c[n]];
}

// ---------------- 7) pooled features ----------------
__global__ void k_counts(const int* __restrict__ nodemap, float* __restrict__ counts) {
  int n = blockIdx.x * 256 + threadIdx.x;
  if (n < N_NODES) atomicAdd(&counts[nodemap[n]], 1.f);
}

__global__ void k_xpool_acc(const float* __restrict__ x, const int* __restrict__ nodemap,
                            float* __restrict__ xp) {
  int idx = blockIdx.x * 256 + threadIdx.x;
  int n = idx >> 7, ch = idx & 127;
  if (n >= N_NODES) return;
  atomicAdd(&xp[(size_t)nodemap[n] * IN_CH + ch], x[idx]);
}

__global__ void k_xpool_div(float* __restrict__ xp, const float* __restrict__ counts,
                            const int* __restrict__ ctr) {
  int idx = blockIdx.x * 256 + threadIdx.x;
  int m = idx >> 7;
  if (m >= ctr[3]) return;
  float c = counts[m];
  xp[idx] = xp[idx] / (c > 0.f ? c : 1.f);
}

// ---------------- 8) pooled edges ----------------
__global__ void k_pooled_edges(const int* __restrict__ ei, const int* __restrict__ nodemap,
                               unsigned long long* __restrict__ tbl,
                               unsigned long long* __restrict__ pk, int* __restrict__ ctr) {
  int i = blockIdx.x * 256 + threadIdx.x;
  if (i >= E_DIR) return;
  int ns = nodemap[ei[i]], nd = nodemap[ei[E_DIR + i]];
  if (ns == nd) return;
  unsigned long long M = (unsigned long long)ctr[3];
  unsigned long long key = (unsigned long long)ns * M + (unsigned long long)nd;
  unsigned int slot = (unsigned int)((key * 0x9E3779B97F4A7C15ull) >> 45) & (HASH2_CAP - 1);
  for (;;) {
    unsigned long long prev = atomicCAS(&tbl[slot], HASH_EMPTY, key);
    if (prev == HASH_EMPTY) { pk[atomicAdd(&ctr[2], 1)] = key; return; }
    if (prev == key) return;
    slot = (slot + 1) & (HASH2_CAP - 1);
  }
}

// ---------------- 9) output packing ----------------
__global__ void k_offsets(int* __restrict__ offs, const int* __restrict__ ctr) {
  if (threadIdx.x != 0 || blockIdx.x != 0) return;
  int M = ctr[3], Ep = ctr[2], Ec = ctr[0];
  offs[0] = M * IN_CH;
  offs[1] = offs[0] + Ep;
  offs[2] = offs[1] + Ep;
  offs[3] = offs[2] + M;
  offs[4] = offs[3] + Ec;
  offs[5] = offs[4] + N_NODES;
}

__global__ void k_pack(float* __restrict__ out, int out_size,
                       const float* __restrict__ xp, const unsigned long long* __restrict__ pk,
                       const int* __restrict__ uniq, const int* __restrict__ batch,
                       const float* __restrict__ scores, const int* __restrict__ nodemap,
                       const int* __restrict__ offs, const int* __restrict__ ctr) {
  int idx = blockIdx.x * 256 + threadIdx.x;
  if (idx >= out_size) return;
  unsigned long long M = (unsigned long long)(ctr[3] > 0 ? ctr[3] : 1);
  float v = 0.f;
  if (idx < offs[0])        v = xp[idx];
  else if (idx < offs[1])  { int j = idx - offs[0]; v = (float)(int)(pk[j] / M); }
  else if (idx < offs[2])  { int j = idx - offs[1]; v = (float)(int)(pk[j] % M); }
  else if (idx < offs[3])  { int m = idx - offs[2]; v = (float)batch[uniq[m]]; }
  else if (idx < offs[4])  { int e = idx - offs[3]; v = scores[e]; }
  else if (idx < offs[5])  { int n = idx - offs[4]; v = (float)nodemap[n]; }
  out[idx] = v;
}

// ==================== host launch ====================
extern "C" void kernel_launch(void* const* d_in, const int* in_sizes, int n_in,
                              void* d_out, int out_size, void* d_ws, size_t ws_size,
                              hipStream_t stream) {
  const float* x      = (const float*)d_in[0];
  const float* W1     = (const float*)d_in[1];
  const float* b1     = (const float*)d_in[2];
  const float* gamma1 = (const float*)d_in[3];
  const float* beta1  = (const float*)d_in[4];
  const float* W2     = (const float*)d_in[5];
  const float* b2     = (const float*)d_in[6];
  const int*   ei     = (const int*)d_in[7];
  const int*   batch  = (const int*)d_in[8];
  (void)in_sizes; (void)n_in; (void)ws_size;

  char* base = (char*)d_ws;
  size_t off = 0;
  auto alloc = [&](size_t bytes) -> void* {
    void* r = base + off;
    off = (off + bytes + 255) & ~(size_t)255;
    return r;
  };
  unsigned long long* hash1 = (unsigned long long*)alloc(HASH1_CAP * 8);
  int*   csrc   = (int*)alloc(E_MAX * 4);
  int*   cdst   = (int*)alloc(E_MAX * 4);
  int*   deg    = (int*)alloc(N_NODES * 4);
  int*   ctr    = (int*)alloc(64 * 4);
  float* dinv   = (float*)alloc(E_MAX * 4);
  unsigned int* Abf = (unsigned int*)alloc((size_t)E_MAX * 64 * 4);
  unsigned int* Bbf = (unsigned int*)alloc(4096 * 4);
  float* H      = (float*)alloc((size_t)E_MAX * HID * 4);
  float* S      = (float*)alloc((size_t)N_NODES * HID * 4);
  float* bnacc  = (float*)alloc(128 * 4);
  float* val    = (float*)alloc(E_MAX * 4);
  float* S2     = (float*)alloc(N_NODES * 4);
  float* scores = (float*)alloc(E_MAX * 4);
  int*   hist   = (int*)alloc(1024 * 4);
  int*   n2c    = (int*)alloc(N_NODES * 4);
  int*   present= (int*)alloc(N_NODES * 4);
  int*   o2n    = (int*)alloc(N_NODES * 4);
  int*   bsums  = (int*)alloc(128 * 4);
  int*   uniq   = (int*)alloc(N_NODES * 4);
  int*   nodemap= (int*)alloc(N_NODES * 4);
  float* counts = (float*)alloc(N_NODES * 4);
  float* xp     = (float*)alloc((size_t)N_NODES * IN_CH * 4);
  unsigned long long* hash2 = (unsigned long long*)alloc(HASH2_CAP * 8);
  unsigned long long* pk    = (unsigned long long*)alloc((size_t)E_DIR * 8);
  int*   offs   = (int*)alloc(16 * 4);

  hipMemsetAsync(hash1, 0xFF, HASH1_CAP * 8, stream);
  hipMemsetAsync(hash2, 0xFF, HASH2_CAP * 8, stream);
  hipMemsetAsync(ctr,    0, 64 * 4, stream);
  hipMemsetAsync(deg,    0, N_NODES * 4, stream);
  hipMemsetAsync(S,      0, (size_t)N_NODES * HID * 4, stream);
  hipMemsetAsync(bnacc,  0, 128 * 4, stream);
  hipMemsetAsync(S2,     0, N_NODES * 4, stream);
  hipMemsetAsync(hist,   0, 1024 * 4, stream);
  hipMemsetAsync(present,0, N_NODES * 4, stream);
  hipMemsetAsync(counts, 0, N_NODES * 4, stream);
  hipMemsetAsync(xp,     0, (size_t)N_NODES * IN_CH * 4, stream);

  const int NB_N    = (N_NODES + 255) / 256;
  const int NB_E    = (E_MAX + 255) / 256;
  const int NB_ED   = (E_DIR + 255) / 256;
  const int NB_EC   = (E_MAX * HID) / 256;
  const int NB_LX   = (E_MAX * 64) / 256;
  const int NB_GEMM = (E_MAX + 127) / 128;
  const int NB_XP   = ((size_t)N_NODES * IN_CH) / 256;

  k_w1bf<<<16, 256, 0, stream>>>(W1, Bbf);
  k_dedup<<<NB_ED, 256, 0, stream>>>(ei, hash1, csrc, cdst, deg, ctr);
  k_dinv<<<NB_E, 256, 0, stream>>>(csrc, cdst, deg, dinv, ctr);
  k_linex<<<NB_LX, 256, 0, stream>>>(x, csrc, cdst, Abf, ctr);
  k_gemm1<<<NB_GEMM, 256, 0, stream>>>(Abf, Bbf, H, ctr);
  k_scatterS<<<NB_EC, 256, 0, stream>>>(H, dinv, csrc, cdst, S, ctr);
  k_conv1<<<NB_EC, 256, 0, stream>>>(H, S, dinv, csrc, cdst, b1, bnacc, ctr);
  k_bn_relu<<<NB_EC, 256, 0, stream>>>(H, bnacc, gamma1, beta1, ctr);
  k_gemv2<<<NB_E, 256, 0, stream>>>(H, W2, val, ctr);
  k_scatterS2<<<NB_E, 256, 0, stream>>>(val, dinv, csrc, cdst, S2, ctr);
  k_score<<<NB_E, 256, 0, stream>>>(val, dinv, csrc, cdst, S2, b2, scores, hist, ctr);
  k_threshold<<<1, 1, 0, stream>>>(hist, ctr);
  k_n2c_init<<<NB_N, 256, 0, stream>>>(n2c);
  k_select<<<NB_E, 256, 0, stream>>>(scores, csrc, cdst, n2c, ctr);
  k_present<<<NB_N, 256, 0, stream>>>(n2c, present);
  k_scan1<<<NB_N, 256, 0, stream>>>(present, o2n, bsums);
  k_scan2<<<1, 1, 0, stream>>>(bsums, NB_N, ctr);
  k_scan3<<<NB_N, 256, 0, stream>>>(present, o2n, bsums, uniq);
  k_nodemap<<<NB_N, 256, 0, stream>>>(n2c, o2n, nodemap);
  k_counts<<<NB_N, 256, 0, stream>>>(nodemap, counts);
  k_xpool_acc<<<NB_XP, 256, 0, stream>>>(x, nodemap, xp);
  k_xpool_div<<<NB_XP, 256, 0, stream>>>(xp, counts, ctr);
  k_pooled_edges<<<NB_ED, 256, 0, stream>>>(ei, nodemap, hash2, pk, ctr);
  k_offsets<<<1, 1, 0, stream>>>(offs, ctr);
  k_pack<<<(out_size + 255) / 256, 256, 0, stream>>>((float*)d_out, out_size, xp, pk,
                                                     uniq, batch, scores, nodemap, offs, ctr);
}
